// SimpleDifferentialGNN_29540785062580
// MI455X (gfx1250) — compile-verified, matrix operand order checked by reading, not run on hardware
//
#include <hip/hip_runtime.h>
#include <hip/hip_bf16.h>

typedef __attribute__((ext_vector_type(16))) __bf16 v16bf;
typedef __attribute__((ext_vector_type(2)))  __bf16 v2bf;
typedef __attribute__((ext_vector_type(8)))  float  v8f;

#define HDIM 128
#define DDIM 64
#define GNUM 64
#define LNUM 3
#define EPSV 1e-5f

// ---------------- utility kernels ----------------

__global__ void fill_kernel(float* p, long long n, float v) {
  long long i = (long long)blockIdx.x * blockDim.x + threadIdx.x;
  if (i < n) p[i] = v;
}

__global__ void deg_count_kernel(const int* __restrict__ dst, float* deg, int E) {
  int e = blockIdx.x * blockDim.x + threadIdx.x;
  if (e < E) unsafeAtomicAdd(&deg[dst[e]], 1.0f);
}

__global__ void rsqrt_kernel(float* p, int n) {
  int i = blockIdx.x * blockDim.x + threadIdx.x;
  if (i < n) p[i] = rsqrtf(p[i]);
}

__global__ void count_nodes_kernel(const int* __restrict__ batch, float* cnt, int n) {
  int i = blockIdx.x * blockDim.x + threadIdx.x;
  if (i < n) unsafeAtomicAdd(&cnt[batch[i]], 1.0f);
}

// ---------------- bf16 WMMA GEMM: C[M,128] = A[M,K] @ W[K,128] (+bias) ----------------
// One wave per 16-row strip of C; the wave keeps 8 f32 accumulators (one per 16-col
// tile) so the streamed A fragment is loaded ONCE per K-step and reused by 8
// v_wmma_f32_16x16x32_bf16 issues. Weights are staged per-block into LDS as bf16,
// transposed (sW[n*K+k]) so each B fragment k-pair is a single aligned ds_load_b32.
// Fragment layouts per CDNA5 ISA 7.12.2 (wave32). Requires M%16==0, K%32==0.

__global__ __launch_bounds__(256) void gemm_bf16_wmma(
    const float* __restrict__ A, const float* __restrict__ W,
    const float* __restrict__ bias, float* __restrict__ C,
    int M, int K) {
  __shared__ __bf16 sW[HDIM * HDIM];          // transposed: sW[n*K + k], 32KB max
  const int tid = threadIdx.x;

  // cooperative stage: global row-major fp32 -> LDS transposed bf16
  for (int i = tid; i < K * HDIM; i += 256) {
    int k = i >> 7, n = i & 127;              // i = k*128 + n (coalesced read)
    sW[n * K + k] = (__bf16)W[i];
  }
  __syncthreads();                            // uniform: before any early-out

  const int strip = blockIdx.x * 8 + (tid >> 5);
  const int lane  = tid & 31;
  if (strip >= (M >> 4)) return;              // wave-uniform: EXEC all-ones below
  const int mrow = strip * 16 + (lane & 15);  // A row for this lane
  const int half = lane >> 4;                 // selects K octet
  const int nlo  = lane & 15;

  v8f zero = {};
  v8f acc[8];
#pragma unroll
  for (int t = 0; t < 8; ++t) acc[t] = zero;

  for (int k0 = 0; k0 < K; k0 += 32) {
    // A fragment: 8 float2 loads (two runs of 4), cvt to bf16
    v16bf a;
#pragma unroll
    for (int v = 0; v < 8; ++v) {
      int kb = k0 + ((v >> 2) << 4) + (half << 3) + ((v & 3) << 1);
      float2 p = *(const float2*)&A[(long long)mrow * K + kb];
      a[2 * v]     = (__bf16)p.x;
      a[2 * v + 1] = (__bf16)p.y;
    }
    if (k0 + 32 < K)                          // gfx1250 global_prefetch of next slab
      __builtin_prefetch(&A[(long long)mrow * K + k0 + 32], 0, 1);

#pragma unroll
    for (int t = 0; t < 8; ++t) {
      v16bf b;
#pragma unroll
      for (int v = 0; v < 8; ++v) {
        int kb = k0 + ((v >> 2) << 4) + (half << 3) + ((v & 3) << 1);
        v2bf p = *(const v2bf*)&sW[(t * 16 + nlo) * K + kb];
        b[2 * v]     = p.x;
        b[2 * v + 1] = p.y;
      }
      acc[t] = __builtin_amdgcn_wmma_f32_16x16x32_bf16(
          false, a, false, b, (short)0, acc[t], false, false);
    }
  }

  const int rowBase = strip * 16 + (half << 3);  // D: VGPR r -> row r (+8 hi half)
#pragma unroll
  for (int t = 0; t < 8; ++t) {
    int col = t * 16 + nlo;
    float bv = bias ? bias[col] : 0.0f;
#pragma unroll
    for (int r = 0; r < 8; ++r)
      C[(long long)(rowBase + r) * HDIM + col] = acc[t][r] + bv;
  }
}

// ---------------- GCN aggregation ----------------
// out[i] = dinv[i]^2 * xw[i] + bias   (self loop term)
__global__ void agg_init_kernel(const float* __restrict__ xw, const float* __restrict__ dinv,
                                const float* __restrict__ bias, float* out, long long total) {
  long long i = (long long)blockIdx.x * blockDim.x + threadIdx.x;
  if (i >= total) return;
  int node = (int)(i >> 7), c = (int)(i & 127);
  float di = dinv[node];
  out[i] = xw[i] * di * di + bias[c];
}

// out[dst] += dinv[src]*dinv[dst] * xw[src]  (thread = one edge x 4 channels)
__global__ void agg_edges_kernel(const float* __restrict__ xw, const int* __restrict__ src,
                                 const int* __restrict__ dst, const float* __restrict__ dinv,
                                 float* out, int E) {
  long long t = (long long)blockIdx.x * blockDim.x + threadIdx.x;
  long long total = (long long)E * 32;
  if (t >= total) return;
  int e  = (int)(t >> 5);
  int cg = (int)(t & 31);
  int s = src[e], d = dst[e];
  float nrm = dinv[s] * dinv[d];
  const float4* x4 = (const float4*)(xw + (long long)s * HDIM);
  float4 v = x4[cg];
  float* o = out + (long long)d * HDIM + cg * 4;
  unsafeAtomicAdd(o + 0, v.x * nrm);
  unsafeAtomicAdd(o + 1, v.y * nrm);
  unsafeAtomicAdd(o + 2, v.z * nrm);
  unsafeAtomicAdd(o + 3, v.w * nrm);
}

// ---------------- graph LayerNorm (mode='graph') ----------------
__global__ void ln_sum_kernel(const float* __restrict__ h, const int* __restrict__ batch,
                              float* gsum, int n) {
  int node = blockIdx.x * (blockDim.x >> 5) + (threadIdx.x >> 5);
  int lane = threadIdx.x & 31;
  if (node >= n) return;
  const float4* r4 = (const float4*)(h + (long long)node * HDIM);
  float4 v = r4[lane];
  float s = v.x + v.y + v.z + v.w;
  for (int off = 16; off > 0; off >>= 1) s += __shfl_xor(s, off, 32);
  if (lane == 0) unsafeAtomicAdd(&gsum[batch[node]], s);
}

__global__ void ln_mean_fin(float* sum, const float* cnt, int g) {
  int i = blockIdx.x * blockDim.x + threadIdx.x;
  if (i < g) sum[i] /= (cnt[i] * (float)HDIM);
}

__global__ void ln_var_kernel(const float* __restrict__ h, const int* __restrict__ batch,
                              const float* __restrict__ mean, float* gvar, int n) {
  int node = blockIdx.x * (blockDim.x >> 5) + (threadIdx.x >> 5);
  int lane = threadIdx.x & 31;
  if (node >= n) return;
  float m = mean[batch[node]];
  const float4* r4 = (const float4*)(h + (long long)node * HDIM);
  float4 v = r4[lane];
  float a = v.x - m, b = v.y - m, c = v.z - m, d = v.w - m;
  float s = a * a + b * b + c * c + d * d;
  for (int off = 16; off > 0; off >>= 1) s += __shfl_xor(s, off, 32);
  if (lane == 0) unsafeAtomicAdd(&gvar[batch[node]], s);
}

__global__ void ln_inv_fin(float* var, const float* cnt, int g) {
  int i = blockIdx.x * blockDim.x + threadIdx.x;
  if (i < g) var[i] = rsqrtf(var[i] / (cnt[i] * (float)HDIM) + EPSV);
}

__global__ void ln_apply_kernel(float* h, const int* __restrict__ batch,
                                const float* __restrict__ mean, const float* __restrict__ inv,
                                const float* __restrict__ w, const float* __restrict__ b,
                                int do_relu, long long total) {
  long long i = (long long)blockIdx.x * blockDim.x + threadIdx.x;
  if (i >= total) return;
  int node = (int)(i >> 7), c = (int)(i & 127);
  int g = batch[node];
  float v = (h[i] - mean[g]) * inv[g] * w[c] + b[c];
  h[i] = do_relu ? fmaxf(v, 0.0f) : v;
}

// global_add_pool
__global__ void pool_kernel(const float* __restrict__ h, const int* __restrict__ batch,
                            float* pool, long long total) {
  long long i = (long long)blockIdx.x * blockDim.x + threadIdx.x;
  if (i >= total) return;
  int node = (int)(i >> 7), c = (int)(i & 127);
  unsafeAtomicAdd(&pool[(long long)batch[node] * HDIM + c], h[i]);
}

// ---------------- MLP head (tiny: G=64 rows) ----------------
__global__ void head1_kernel(const float* __restrict__ poolK, const float* __restrict__ poolD,
                             const float* __restrict__ W1, const float* __restrict__ b1,
                             float* h1) {
  int g = blockIdx.x, c = threadIdx.x;           // grid GNUM, block HDIM
  float acc = b1[c];
  for (int k = 0; k < HDIM; ++k)
    acc += poolK[g * HDIM + k] * W1[k * HDIM + c];
  for (int k = 0; k < HDIM; ++k)
    acc += poolD[g * HDIM + k] * W1[(HDIM + k) * HDIM + c];
  h1[g * HDIM + c] = fmaxf(acc, 0.0f);
}

__global__ void head2_kernel(const float* __restrict__ h1, const float* __restrict__ W2,
                             const float* __restrict__ b2, float* h2) {
  int g = blockIdx.x, c = threadIdx.x;           // grid GNUM, block HDIM/2
  float acc = b2[c];
  for (int k = 0; k < HDIM; ++k)
    acc += h1[g * HDIM + k] * W2[k * (HDIM / 2) + c];
  h2[g * (HDIM / 2) + c] = fmaxf(acc, 0.0f);
}

__global__ void head3_kernel(const float* __restrict__ h2, const float* __restrict__ W3,
                             const float* __restrict__ b3, float* out) {
  int g = threadIdx.x;                           // one block, GNUM threads
  if (g >= GNUM) return;
  float acc = b3[0];
  for (int k = 0; k < HDIM / 2; ++k)
    acc += h2[g * (HDIM / 2) + k] * W3[k];
  out[g] = acc;
}

// ---------------- host orchestration ----------------

static inline int nblk(long long n, int t) { return (int)((n + t - 1) / t); }

extern "C" void kernel_launch(void* const* d_in, const int* in_sizes, int n_in,
                              void* d_out, int out_size, void* d_ws, size_t ws_size,
                              hipStream_t stream) {
  const float* kx   = (const float*)d_in[0];
  const int*   kei  = (const int*)  d_in[1];
  const int*   kb   = (const int*)  d_in[2];
  const float* dx   = (const float*)d_in[3];
  const int*   dei  = (const int*)  d_in[4];
  const int*   db   = (const int*)  d_in[5];
  // d_in[6] = pragma_count (unused by reference math)
  const float* encW  = (const float*)d_in[7];
  const float* encB  = (const float*)d_in[8];
  const float* convW = (const float*)d_in[9];
  const float* convB = (const float*)d_in[10];
  const float* lnw   = (const float*)d_in[11];
  const float* lnb   = (const float*)d_in[12];
  const float* W1    = (const float*)d_in[13];
  const float* b1    = (const float*)d_in[14];
  const float* W2    = (const float*)d_in[15];
  const float* b2    = (const float*)d_in[16];
  const float* W3    = (const float*)d_in[17];
  const float* b3    = (const float*)d_in[18];
  float* out = (float*)d_out;

  const int N = in_sizes[0] / DDIM;
  const int E = in_sizes[1] / 2;
  const long long NH = (long long)N * HDIM;

  // workspace layout
  float* ws    = (float*)d_ws;
  float* hbuf  = ws;                 // N*H
  float* tmp   = hbuf + NH;          // N*H
  float* dinv  = tmp + NH;           // N
  float* cnt   = dinv + N;           // G
  float* meanb = cnt + GNUM;         // G
  float* varb  = meanb + GNUM;       // G
  float* poolK = varb + GNUM;        // G*H
  float* poolD = poolK + (long long)GNUM * HDIM;  // G*H
  float* h1    = poolD + (long long)GNUM * HDIM;  // G*H
  float* h2    = h1 + (long long)GNUM * HDIM;     // G*H/2

  const int T = 256;
  const int gemmBlocks = nblk(N / 16, 8);         // 8 strip-waves per block

  auto encode = [&](const float* x, const int* ei, const int* batch, float* pool) {
    const int* src = ei;
    const int* dst = ei + E;

    // degrees (self-loop => deg starts at 1) and D^{-1/2}
    fill_kernel<<<nblk(N, T), T, 0, stream>>>(dinv, N, 1.0f);
    deg_count_kernel<<<nblk(E, T), T, 0, stream>>>(dst, dinv, E);
    rsqrt_kernel<<<nblk(N, T), T, 0, stream>>>(dinv, N);

    // per-graph node counts
    fill_kernel<<<1, GNUM, 0, stream>>>(cnt, GNUM, 0.0f);
    count_nodes_kernel<<<nblk(N, T), T, 0, stream>>>(batch, cnt, N);

    // encoder: h = x @ enc_W + enc_b   (K = 64)
    gemm_bf16_wmma<<<gemmBlocks, T, 0, stream>>>(x, encW, encB, hbuf, N, DDIM);

    for (int l = 0; l < LNUM; ++l) {
      // xw = h @ conv_W[l]             (K = 128)
      gemm_bf16_wmma<<<gemmBlocks, T, 0, stream>>>(hbuf, convW + (long long)l * HDIM * HDIM,
                                                   nullptr, tmp, N, HDIM);
      // h = D^{-1/2}(A+I)D^{-1/2} xw + conv_b[l]
      agg_init_kernel<<<nblk(NH, T), T, 0, stream>>>(tmp, dinv, convB + l * HDIM, hbuf, NH);
      agg_edges_kernel<<<nblk((long long)E * 32, T), T, 0, stream>>>(tmp, src, dst, dinv, hbuf, E);

      // graph layernorm
      fill_kernel<<<1, GNUM, 0, stream>>>(meanb, GNUM, 0.0f);
      ln_sum_kernel<<<nblk(N, 8), T, 0, stream>>>(hbuf, batch, meanb, N);
      ln_mean_fin<<<1, GNUM, 0, stream>>>(meanb, cnt, GNUM);
      fill_kernel<<<1, GNUM, 0, stream>>>(varb, GNUM, 0.0f);
      ln_var_kernel<<<nblk(N, 8), T, 0, stream>>>(hbuf, batch, meanb, varb, N);
      ln_inv_fin<<<1, GNUM, 0, stream>>>(varb, cnt, GNUM);
      ln_apply_kernel<<<nblk(NH, T), T, 0, stream>>>(hbuf, batch, meanb, varb,
                                                     lnw + l * HDIM, lnb + l * HDIM,
                                                     (l < LNUM - 1) ? 1 : 0, NH);
    }

    // global add pool
    fill_kernel<<<nblk((long long)GNUM * HDIM, T), T, 0, stream>>>(pool, (long long)GNUM * HDIM, 0.0f);
    pool_kernel<<<nblk(NH, T), T, 0, stream>>>(hbuf, batch, pool, NH);
  };

  encode(kx, kei, kb, poolK);   // kernel graph
  encode(dx, dei, db, poolD);   // design graph

  // head MLP on [G, 2H] = concat(kernel_repr, design_repr)
  head1_kernel<<<GNUM, HDIM, 0, stream>>>(poolK, poolD, W1, b1, h1);
  head2_kernel<<<GNUM, HDIM / 2, 0, stream>>>(h1, W2, b2, h2);
  head3_kernel<<<1, GNUM, 0, stream>>>(h2, W3, b3, out);
}
